// YOLOLoss_29721173688536
// MI455X (gfx1250) — matrix operand order repaired and checked by reading, not compiled, not run on hardware
//
#include <hip/hip_runtime.h>
#include <cstdint>
#include <cstddef>

// ---------------- problem constants ----------------
constexpr int BATCH  = 32;
constexpr int NGT    = 30;
constexpr int NCLS   = 80;
constexpr int KPAD   = 96;          // classes padded to 3 x 32 for WMMA K
constexpr int A_TOT  = 8400;        // 80*80 + 40*40 + 20*20
constexpr int NTILES = A_TOT / 16;  // 525 anchor tiles
constexpr int KTILES = 3;

typedef __attribute__((ext_vector_type(16))) _Float16 v16h;
typedef __attribute__((ext_vector_type(8)))  float    v8f;

__device__ __forceinline__ float sigf(float x) { return 1.0f / (1.0f + expf(-x)); }

__device__ __forceinline__ float bcef(float x, float t) {
  return fmaxf(x, 0.0f) - x * t + log1pf(expf(-fabsf(x)));
}

// IoU of two cxcywh boxes (matches reference pairwise_iou / iou_elem, eps 1e-16)
__device__ __forceinline__ float iou_pair(float gx, float gy, float gw, float gh,
                                          float px, float py, float pw, float ph) {
  float tlx = fmaxf(gx - 0.5f * gw, px - 0.5f * pw);
  float tly = fmaxf(gy - 0.5f * gh, py - 0.5f * ph);
  float brx = fminf(gx + 0.5f * gw, px + 0.5f * pw);
  float bry = fminf(gy + 0.5f * gh, py + 0.5f * ph);
  float en  = ((tlx < brx) && (tly < bry)) ? 1.0f : 0.0f;
  float inter = (brx - tlx) * (bry - tly) * en;
  return inter / (gw * gh + pw * ph - inter + 1e-16f);
}

__device__ __forceinline__ void anchor_level(int a, int& pos, int& W, int& HW, float& s) {
  if (a < 6400)      { pos = a;        W = 80; HW = 6400; s = 8.0f;  }
  else if (a < 8000) { pos = a - 6400; W = 40; HW = 1600; s = 16.0f; }
  else               { pos = a - 8000; W = 20; HW = 400;  s = 32.0f; }
}

// =====================================================================
// Kernel 1: decode + per-anchor stats + geometry masks + swizzled f16 D
// grid (ceil(A/256), B), block 256
// =====================================================================
__global__ void k_decode(const float* __restrict__ f8, const float* __restrict__ f16,
                         const float* __restrict__ f32v, const float* __restrict__ labels,
                         float4* __restrict__ bboxP, float* __restrict__ objL,
                         float* __restrict__ baseA, unsigned* __restrict__ aok,
                         unsigned* __restrict__ inboth, void* __restrict__ DfragV) {
  _Float16* Dfrag = (_Float16*)DfragV;
  __shared__ float sLab[NGT * 5];
  const int b = blockIdx.y;
  const int tid = threadIdx.x;
  if (tid < NGT * 5) sLab[tid] = labels[(size_t)b * NGT * 5 + tid];
  __syncthreads();
  const int a = blockIdx.x * 256 + tid;
  if (a >= A_TOT) return;

  int pos, W, HW; float s;
  anchor_level(a, pos, W, HW, s);
  const float* f = (a < 6400) ? f8 : (a < 8000) ? f16 : f32v;
  const float* fb = f + (size_t)b * 85 * HW + pos;

  const float gx = (float)(pos % W), gy = (float)(pos / W);
  const float bx = (fb[0] + gx) * s;
  const float by = (fb[(size_t)HW] + gy) * s;
  const float bw = expf(fb[2 * (size_t)HW]) * s;
  const float bh = expf(fb[3 * (size_t)HW]) * s;
  bboxP[(size_t)b * A_TOT + a] = make_float4(bx, by, bw, bh);

  const float ob = fb[4 * (size_t)HW];
  objL[(size_t)b * A_TOT + a] = ob;
  const float osig = sigf(ob);

  // D = log1mp - logp per class, written in this code's contiguous-K B-fragment order:
  // tile (ktile,ntile): lane = N%16 + 16*(K-half), element e = K%16.
  float basev = 0.0f;
  const int ntile = a >> 4, nl = a & 15;
  for (int c = 0; c < KPAD; ++c) {
    float d = 0.0f;
    if (c < NCLS) {
      float x  = fb[(size_t)(5 + c) * HW];
      float p  = sqrtf(sigf(x) * osig);
      float lp = logf(p + 1e-8f);
      float l1 = logf(1.0f - p + 1e-8f);
      basev += l1;
      d = l1 - lp;
    }
    const int kt = c >> 5, kin = c & 31;
    const int lane = nl + ((kin & 16) ? 16 : 0);
    const int e = kin & 15;
    const size_t idx = ((((size_t)b * KTILES + kt) * NTILES + ntile) << 9) + lane * 16 + e;
    Dfrag[idx] = (_Float16)d;
  }
  baseA[(size_t)b * A_TOT + a] = -basev;

  // geometry: in_box / in_ctr bits over all 30 GTs
  const float xc = (gx + 0.5f) * s, yc = (gy + 0.5f) * s;
  const float rr = 2.5f * s;
  unsigned okb = 0u, ibb = 0u;
  for (int g = 0; g < NGT; ++g) {
    const float cx = sLab[g * 5 + 0], cy = sLab[g * 5 + 1];
    const float w2 = 0.5f * sLab[g * 5 + 2], h2 = 0.5f * sLab[g * 5 + 3];
    const bool inbox = (xc > cx - w2) && (xc < cx + w2) && (yc > cy - h2) && (yc < cy + h2);
    const bool inctr = (xc > cx - rr) && (xc < cx + rr) && (yc > cy - rr) && (yc < cy + rr);
    if (inbox || inctr) okb = 1u;
    if (inbox && inctr) ibb |= (1u << g);
  }
  aok[(size_t)b * A_TOT + a]    = okb;
  inboth[(size_t)b * A_TOT + a] = ibb;
}

// =====================================================================
// Kernel 2: cost matrix via WMMA (one-hot x D) + IoU cost + penalties
// grid (66, 2, B), block 256 (8 waves, one 16x16 tile per wave)
// =====================================================================
__global__ void k_cost(const float* __restrict__ labels, const float4* __restrict__ bboxP,
                       const float* __restrict__ baseA, const unsigned* __restrict__ aok,
                       const unsigned* __restrict__ inboth, const void* __restrict__ DfragV,
                       float* __restrict__ costM) {
  const _Float16* Dfrag = (const _Float16*)DfragV;
  const int b = blockIdx.z, gtile = blockIdx.y;
  __shared__ float sBox[16][4];
  __shared__ int   sCls[16];
  const int tid = threadIdx.x;
  if (tid < 16) {
    const int g = gtile * 16 + tid;
    if (g < NGT) {
      const float* L = labels + ((size_t)b * NGT + g) * 5;
      sBox[tid][0] = L[0]; sBox[tid][1] = L[1]; sBox[tid][2] = L[2]; sBox[tid][3] = L[3];
      sCls[tid] = (int)L[4];
    } else {
      sBox[tid][0] = sBox[tid][1] = sBox[tid][2] = sBox[tid][3] = 0.0f;
      sCls[tid] = -1;   // never matches any K -> zero one-hot row
    }
  }
  __syncthreads();

  const int wid = tid >> 5, lane = tid & 31;
  const int nt = blockIdx.x * 8 + wid;          // wave-uniform
  if (nt >= NTILES) return;
  const int half = lane >> 4, m = lane & 15;
  const int myCls = sCls[m];

  // one-hot A fragments (contiguous K order, same as D: K = e + 16*half)
  union { v16h v; _Float16 h[16]; } afr[KTILES];
  #pragma unroll
  for (int kt = 0; kt < KTILES; ++kt)
    #pragma unroll
    for (int e = 0; e < 16; ++e) {
      const int K = kt * 32 + e + 16 * half;
      afr[kt].h[e] = (myCls == K) ? (_Float16)1.0f : (_Float16)0.0f;
    }

  union { v16h v; float4 q[2]; } bfr;
  union { v8f v; float f[8]; } cfr;
  #pragma unroll
  for (int k = 0; k < 8; ++k) cfr.f[k] = 0.0f;

  #pragma unroll
  for (int kt = 0; kt < KTILES; ++kt) {
    const size_t base = ((((size_t)b * KTILES + kt) * NTILES + nt) << 9) + (size_t)lane * 16;
    if (kt < KTILES - 1)
      __builtin_prefetch((const void*)(Dfrag + base + ((size_t)NTILES << 9)), 0, 0);
    const float4* src = (const float4*)(Dfrag + base);
    bfr.q[0] = src[0];
    bfr.q[1] = src[1];
    cfr.v = __builtin_amdgcn_wmma_f32_16x16x32_f16(
        false, afr[kt].v, false, bfr.v, (short)0, cfr.v, false, false);
  }

  const int a = nt * 16 + m;                    // C layout: N = lane%16
  const float4   bp    = bboxP[(size_t)b * A_TOT + a];
  const float    basev = baseA[(size_t)b * A_TOT + a];
  const unsigned okv   = aok[(size_t)b * A_TOT + a];
  const unsigned ibb   = inboth[(size_t)b * A_TOT + a];

  #pragma unroll
  for (int r = 0; r < 8; ++r) {                 // C layout: M = r + 8*half
    const int g = gtile * 16 + r + 8 * half;
    if (g >= NGT) continue;
    const int gm = r + 8 * half;
    const float iou = iou_pair(sBox[gm][0], sBox[gm][1], sBox[gm][2], sBox[gm][3],
                               bp.x, bp.y, bp.z, bp.w);
    const float ioum  = okv ? iou : 0.0f;
    const float icost = -logf(ioum + 1e-8f);
    const float cost  = cfr.f[r] + basev + 3.0f * icost
                      + (((ibb >> g) & 1u) ? 0.0f : 100000.0f)
                      + (okv ? 0.0f : 1000000000.0f);
    costM[((size_t)b * NGT + g) * A_TOT + a] = cost;
  }
}

// =====================================================================
// Kernel 3: per-(b,g) row: top-10 masked IoUs -> dyn_k; top-10 (cost,idx)
// pairs -> lexicographic threshold pair.  grid (30, B), block 256
// =====================================================================
__device__ __forceinline__ bool lex_less(float v1, int i1, float v2, int i2) {
  return (v1 < v2) || (v1 == v2 && i1 < i2);
}

__global__ void k_topk(const float* __restrict__ labels, const float4* __restrict__ bboxP,
                       const unsigned* __restrict__ aok, const float* __restrict__ costM,
                       float* __restrict__ thrC, int* __restrict__ thrI) {
  const int g = blockIdx.x, b = blockIdx.y;
  const int tid = threadIdx.x;
  __shared__ float sV[256 * 10];
  __shared__ int   sI[256 * 10];
  __shared__ int   sDynK;

  const float* L = labels + ((size_t)b * NGT + g) * 5;
  const float gx = L[0], gy = L[1], gw = L[2], gh = L[3];

  // ---- pass 1: masked IoU, 10 largest (desc) ----
  float loc[10];
  #pragma unroll
  for (int k = 0; k < 10; ++k) loc[k] = -1.0f;
  for (int a = tid; a < A_TOT; a += 256) {
    const float4 p = bboxP[(size_t)b * A_TOT + a];
    float iou = aok[(size_t)b * A_TOT + a]
              ? iou_pair(gx, gy, gw, gh, p.x, p.y, p.z, p.w) : 0.0f;
    if (iou > loc[9]) {
      int j = 9;
      while (j > 0 && iou > loc[j - 1]) { loc[j] = loc[j - 1]; --j; }
      loc[j] = iou;
    }
  }
  #pragma unroll
  for (int k = 0; k < 10; ++k) sV[tid * 10 + k] = loc[k];
  for (int s = 128; s > 0; s >>= 1) {
    __syncthreads();
    if (tid < s) {
      float aL[10], bL[10], o[10];
      #pragma unroll
      for (int k = 0; k < 10; ++k) { aL[k] = sV[tid * 10 + k]; bL[k] = sV[(tid + s) * 10 + k]; }
      int ia = 0, ib = 0;
      #pragma unroll
      for (int k = 0; k < 10; ++k) o[k] = (aL[ia] >= bL[ib]) ? aL[ia++] : bL[ib++];
      #pragma unroll
      for (int k = 0; k < 10; ++k) sV[tid * 10 + k] = o[k];
    }
  }
  __syncthreads();
  if (tid == 0) {
    float su = 0.0f;
    #pragma unroll
    for (int k = 0; k < 10; ++k) su += sV[k];
    int dk = (int)su;                // trunc toward zero, matches astype(int32)
    if (dk < 1) dk = 1;
    if (dk > 10) dk = 10;
    sDynK = dk;
  }
  __syncthreads();

  // ---- pass 2: (cost, idx) 10 smallest (lexicographic asc) ----
  float lv[10]; int li[10];
  #pragma unroll
  for (int k = 0; k < 10; ++k) { lv[k] = 3.402823466e38f; li[k] = 0x7fffffff; }
  const float* crow = costM + ((size_t)b * NGT + g) * A_TOT;
  for (int a = tid; a < A_TOT; a += 256) {
    const float c = crow[a];
    if (lex_less(c, a, lv[9], li[9])) {
      int j = 9;
      while (j > 0 && lex_less(c, a, lv[j - 1], li[j - 1])) {
        lv[j] = lv[j - 1]; li[j] = li[j - 1]; --j;
      }
      lv[j] = c; li[j] = a;
    }
  }
  #pragma unroll
  for (int k = 0; k < 10; ++k) { sV[tid * 10 + k] = lv[k]; sI[tid * 10 + k] = li[k]; }
  for (int s = 128; s > 0; s >>= 1) {
    __syncthreads();
    if (tid < s) {
      float aV[10], bV[10], oV[10]; int aI[10], bI[10], oI[10];
      #pragma unroll
      for (int k = 0; k < 10; ++k) {
        aV[k] = sV[tid * 10 + k]; aI[k] = sI[tid * 10 + k];
        bV[k] = sV[(tid + s) * 10 + k]; bI[k] = sI[(tid + s) * 10 + k];
      }
      int ia = 0, ib = 0;
      #pragma unroll
      for (int k = 0; k < 10; ++k) {
        if (lex_less(aV[ia], aI[ia], bV[ib], bI[ib])) { oV[k] = aV[ia]; oI[k] = aI[ia]; ++ia; }
        else                                          { oV[k] = bV[ib]; oI[k] = bI[ib]; ++ib; }
      }
      #pragma unroll
      for (int k = 0; k < 10; ++k) { sV[tid * 10 + k] = oV[k]; sI[tid * 10 + k] = oI[k]; }
    }
  }
  __syncthreads();
  if (tid == 0) {
    const int dk = sDynK;
    thrC[b * NGT + g] = sV[dk - 1];
    thrI[b * NGT + g] = sI[dk - 1];
  }
}

// =====================================================================
// Kernel 4: per-anchor match resolve.  grid (ceil(A/256), B), block 256
// =====================================================================
__global__ void k_resolve(const float* __restrict__ labels, const float4* __restrict__ bboxP,
                          const unsigned* __restrict__ aok, const float* __restrict__ costM,
                          const float* __restrict__ thrC, const int* __restrict__ thrI,
                          float* __restrict__ fgA, int* __restrict__ mgtA,
                          float* __restrict__ piouA) {
  __shared__ float sC[NGT]; __shared__ int sIx[NGT]; __shared__ float sLab[NGT * 5];
  const int b = blockIdx.y, tid = threadIdx.x;
  if (tid < NGT) { sC[tid] = thrC[b * NGT + tid]; sIx[tid] = thrI[b * NGT + tid]; }
  if (tid < NGT * 5) sLab[tid] = labels[(size_t)b * NGT * 5 + tid];
  __syncthreads();
  const int a = blockIdx.x * 256 + tid;
  if (a >= A_TOT) return;

  const unsigned okv = aok[(size_t)b * A_TOT + a];
  int cnt = 0, firstg = 0, bg = 0;
  float bc = 3.402823466e38f;
  bool any = false;
  for (int g = 0; g < NGT; ++g) {
    const float c = costM[((size_t)b * NGT + g) * A_TOT + a];
    if (c < bc) { bc = c; bg = g; }              // first-min tie-break (strict <)
    if (okv && (c < sC[g] || (c == sC[g] && a <= sIx[g]))) {
      ++cnt;
      if (!any) { firstg = g; any = true; }
    }
  }
  float fgv = 0.0f; int mg = 0; float piou = 0.0f;
  if (cnt > 1)       { mg = bg;     fgv = 1.0f; }
  else if (cnt == 1) { mg = firstg; fgv = 1.0f; }
  if (fgv > 0.0f) {
    const float4 p = bboxP[(size_t)b * A_TOT + a];
    piou = iou_pair(sLab[mg * 5], sLab[mg * 5 + 1], sLab[mg * 5 + 2], sLab[mg * 5 + 3],
                    p.x, p.y, p.z, p.w);
  }
  fgA[(size_t)b * A_TOT + a]  = fgv;
  mgtA[(size_t)b * A_TOT + a] = mg;
  piouA[(size_t)b * A_TOT + a] = piou;
}

// =====================================================================
// Kernel 5: per-anchor loss contributions -> per-block partial sums
// grid (33, B), block 256.  partials[blk][4] = {iou, obj, cls, num_fg}
// =====================================================================
__global__ void k_loss(const float* __restrict__ f8, const float* __restrict__ f16,
                       const float* __restrict__ f32v, const float* __restrict__ labels,
                       const float4* __restrict__ bboxP, const float* __restrict__ objL,
                       const float* __restrict__ fgA, const int* __restrict__ mgtA,
                       const float* __restrict__ piouA, float* __restrict__ partials) {
  const int b = blockIdx.y, tid = threadIdx.x;
  const int a = blockIdx.x * 256 + tid;
  float li = 0.0f, lo = 0.0f, lc = 0.0f, nf = 0.0f;
  if (a < A_TOT) {
    const float t = fgA[(size_t)b * A_TOT + a];
    const float x = objL[(size_t)b * A_TOT + a];
    lo = bcef(x, t);
    nf = t;
    if (t > 0.0f) {
      const int mg = mgtA[(size_t)b * A_TOT + a];
      const float* L = labels + ((size_t)b * NGT + mg) * 5;
      const float4 p = bboxP[(size_t)b * A_TOT + a];
      const float iou = iou_pair(L[0], L[1], L[2], L[3], p.x, p.y, p.z, p.w);
      li = 1.0f - iou * iou;
      const int gcls = (int)L[4];
      const float pi = piouA[(size_t)b * A_TOT + a];
      int pos, W, HW; float s;
      anchor_level(a, pos, W, HW, s);
      const float* f = (a < 6400) ? f8 : (a < 8000) ? f16 : f32v;
      const float* fb = f + (size_t)b * 85 * HW + pos;
      for (int c = 0; c < NCLS; ++c) {
        const float xl = fb[(size_t)(5 + c) * HW];
        lc += bcef(xl, (c == gcls) ? pi : 0.0f);
      }
    }
  }
  __shared__ float s0[256], s1[256], s2[256], s3[256];
  s0[tid] = li; s1[tid] = lo; s2[tid] = lc; s3[tid] = nf;
  for (int s = 128; s > 0; s >>= 1) {
    __syncthreads();
    if (tid < s) {
      s0[tid] += s0[tid + s]; s1[tid] += s1[tid + s];
      s2[tid] += s2[tid + s]; s3[tid] += s3[tid + s];
    }
  }
  if (tid == 0) {
    const int blk = blockIdx.y * gridDim.x + blockIdx.x;
    partials[blk * 4 + 0] = s0[0]; partials[blk * 4 + 1] = s1[0];
    partials[blk * 4 + 2] = s2[0]; partials[blk * 4 + 3] = s3[0];
  }
}

// Kernel 6: deterministic final reduction + scalar loss
__global__ void k_final(const float* __restrict__ partials, int nblk, float* __restrict__ out) {
  const int tid = threadIdx.x;
  float a0 = 0, a1 = 0, a2 = 0, a3 = 0;
  for (int i = tid; i < nblk; i += 256) {
    a0 += partials[i * 4 + 0]; a1 += partials[i * 4 + 1];
    a2 += partials[i * 4 + 2]; a3 += partials[i * 4 + 3];
  }
  __shared__ float s0[256], s1[256], s2[256], s3[256];
  s0[tid] = a0; s1[tid] = a1; s2[tid] = a2; s3[tid] = a3;
  for (int s = 128; s > 0; s >>= 1) {
    __syncthreads();
    if (tid < s) {
      s0[tid] += s0[tid + s]; s1[tid] += s1[tid + s];
      s2[tid] += s2[tid + s]; s3[tid] += s3[tid + s];
    }
  }
  if (tid == 0) {
    const float nfg = fmaxf(s3[0], 1.0f);
    out[0] = (5.0f * s0[0] + 0.1f * s1[0] + s2[0]) / nfg;
  }
}

// =====================================================================
extern "C" void kernel_launch(void* const* d_in, const int* in_sizes, int n_in,
                              void* d_out, int out_size, void* d_ws, size_t ws_size,
                              hipStream_t stream) {
  const float* f8     = (const float*)d_in[0];
  const float* f16    = (const float*)d_in[1];
  const float* f32v   = (const float*)d_in[2];
  const float* labels = (const float*)d_in[3];
  float* out = (float*)d_out;

  char* w = (char*)d_ws;
  size_t off = 0;
  auto carve = [&](size_t bytes) -> char* {
    char* p = w + off;
    off = (off + bytes + 255) & ~(size_t)255;
    return p;
  };
  float4*   bboxP  = (float4*)  carve((size_t)BATCH * A_TOT * sizeof(float4));
  float*    objL   = (float*)   carve((size_t)BATCH * A_TOT * sizeof(float));
  float*    baseA  = (float*)   carve((size_t)BATCH * A_TOT * sizeof(float));
  unsigned* aok    = (unsigned*)carve((size_t)BATCH * A_TOT * sizeof(unsigned));
  unsigned* inboth = (unsigned*)carve((size_t)BATCH * A_TOT * sizeof(unsigned));
  void*     Dfrag  = (void*)    carve((size_t)BATCH * KTILES * NTILES * 512 * 2); // f16
  float*    costM  = (float*)   carve((size_t)BATCH * NGT * A_TOT * sizeof(float));
  float*    thrC   = (float*)   carve((size_t)BATCH * NGT * sizeof(float));
  int*      thrI   = (int*)     carve((size_t)BATCH * NGT * sizeof(int));
  float*    fgA    = (float*)   carve((size_t)BATCH * A_TOT * sizeof(float));
  int*      mgtA   = (int*)     carve((size_t)BATCH * A_TOT * sizeof(int));
  float*    piouA  = (float*)   carve((size_t)BATCH * A_TOT * sizeof(float));
  const int ABLK = (A_TOT + 255) / 256;          // 33
  float*    parts  = (float*)   carve((size_t)ABLK * BATCH * 4 * sizeof(float));

  dim3 blk(256);
  dim3 grdA(ABLK, BATCH);
  k_decode<<<grdA, blk, 0, stream>>>(f8, f16, f32v, labels, bboxP, objL, baseA,
                                     aok, inboth, Dfrag);

  dim3 grdC((NTILES + 7) / 8, 2, BATCH);         // 66 x 2 x 32, 8 waves/block
  k_cost<<<grdC, blk, 0, stream>>>(labels, bboxP, baseA, aok, inboth, Dfrag, costM);

  dim3 grdT(NGT, BATCH);
  k_topk<<<grdT, blk, 0, stream>>>(labels, bboxP, aok, costM, thrC, thrI);

  k_resolve<<<grdA, blk, 0, stream>>>(labels, bboxP, aok, costM, thrC, thrI,
                                      fgA, mgtA, piouA);

  k_loss<<<grdA, blk, 0, stream>>>(f8, f16, f32v, labels, bboxP, objL,
                                   fgA, mgtA, piouA, parts);

  k_final<<<dim3(1), blk, 0, stream>>>(parts, ABLK * BATCH, out);
}
